// NetGCN_15324443312369
// MI455X (gfx1250) — matrix operand order repaired and checked by reading, not compiled, not run on hardware
//
#include <hip/hip_runtime.h>
#include <hip/hip_bf16.h>

typedef __attribute__((ext_vector_type(8)))  _Float16 v8h;
typedef __attribute__((ext_vector_type(16))) _Float16 v16h;
typedef __attribute__((ext_vector_type(8)))  float    v8f;

#define N_OBJ   20000
#define N_ROOM  500
#define N_ATTR  2000
#define IN_F    300
#define HDIM    512
#define E_OO    640000
#define E_RO    50000
#define E_AO    100000
#define NBATCH  8

// ---------------- utility kernels ----------------

__global__ void fill_kernel(float* __restrict__ p, float v, long n) {
    long i = (long)blockIdx.x * blockDim.x + threadIdx.x;
    if (i < n) p[i] = v;
}

__global__ void degree_kernel(const int* __restrict__ src, const int* __restrict__ dst,
                              float* __restrict__ dsrc, float* __restrict__ ddst, int E) {
    int e = blockIdx.x * blockDim.x + threadIdx.x;
    if (e < E) {
        atomicAdd(&dsrc[src[e]], 1.0f);
        atomicAdd(&ddst[dst[e]], 1.0f);
    }
}

// in-place deg -> clip(deg,1)^-1/2
__global__ void norm_kernel(float* __restrict__ p, long n) {
    long i = (long)blockIdx.x * blockDim.x + threadIdx.x;
    if (i < n) p[i] = rsqrtf(fmaxf(p[i], 1.0f));
}

// one block per edge; lanes stride over features
__global__ void scatter_kernel(const float* __restrict__ feat,
                               const int* __restrict__ src, const int* __restrict__ dst,
                               const float* __restrict__ nsrc,
                               float* __restrict__ agg, int F) {
    int e = blockIdx.x;
    int s = src[e], d = dst[e];
    float w = nsrc[s];
    const float* frow = feat + (long)s * F;
    float* arow = agg + (long)d * F;
    for (int f = threadIdx.x; f < F; f += blockDim.x)
        atomicAdd(&arow[f], frow[f] * w);
}

// broadcast slice 0 of d_out to slices 1..copies-1
__global__ void bcast_kernel(float* __restrict__ out, long SL, int copies) {
    long i = (long)blockIdx.x * blockDim.x + threadIdx.x;
    if (i < SL) {
        float v = out[i];
        for (int b = 1; b < copies; ++b) out[b * SL + i] = v;
    }
}

// ---------------- f32 -> f16 hi/lo split conversions ----------------

// A[M,K] f32 (row i scaled by ndst[i]*scale) -> Ahi/Alo [M,Kp] f16, zero-padded
__global__ void convA_kernel(const float* __restrict__ A, const float* __restrict__ ndst,
                             float scale, _Float16* __restrict__ Ahi, _Float16* __restrict__ Alo,
                             int K, int Kp, long total) {
    long i = (long)blockIdx.x * blockDim.x + threadIdx.x;
    if (i >= total) return;
    int r = (int)(i / Kp);
    int c = (int)(i % Kp);
    float v = (c < K) ? A[(long)r * K + c] * ndst[r] * scale : 0.0f;
    _Float16 h = (_Float16)v;
    Ahi[i] = h;
    Alo[i] = (_Float16)(v - (float)h);
}

// W[K,N] f32 -> packed B fragments in WMMA 16-bit lane layout, zero-padded in K.
// Fragment index: ((kt * (N/16) + nt) * 32 + lane) * 16 + j
// holds W[kt*32 + 16*(j/8) + 8*(lane/16) + j%8][nt*16 + lane%16]
__global__ void convB_kernel(const float* __restrict__ W,
                             _Float16* __restrict__ Bhi, _Float16* __restrict__ Blo,
                             int K, int N, long total) {
    long i = (long)blockIdx.x * blockDim.x + threadIdx.x;
    if (i >= total) return;
    int j    = (int)(i & 15);
    int lane = (int)((i >> 4) & 31);
    long tile = i >> 9;
    int ntiles16 = N >> 4;
    int kt = (int)(tile / ntiles16);
    int nt = (int)(tile % ntiles16);
    int k   = kt * 32 + ((j >> 3) << 4) + ((lane >> 4) << 3) + (j & 7);
    int col = nt * 16 + (lane & 15);
    float v = (k < K) ? W[(long)k * N + col] : 0.0f;
    _Float16 h = (_Float16)v;
    Bhi[i] = h;
    Blo[i] = (_Float16)(v - (float)h);
}

// ---------------- split-f16 WMMA GEMM ----------------
// C[M,N] (op) = A[M,Kp]@B[Kp,N] + bias[N], A/B given as f16 hi+lo pairs.
// mode 0: C = relu(x)   mode 1: C += relu(x)   mode 2: C = x
// 256 threads = 8 waves; each wave computes a 16x64 strip with 4 f32 accums,
// 3 x v_wmma_f32_16x16x32_f16 per (K-tile, col-tile): hi*hi + hi*lo + lo*hi.
__global__ __launch_bounds__(256) void gemm_wmma_f16x2(
    const _Float16* __restrict__ Ahi, const _Float16* __restrict__ Alo, int Kp,
    const _Float16* __restrict__ Bhi, const _Float16* __restrict__ Blo,
    const float* __restrict__ bias, float* __restrict__ C,
    int M, int N, int mode)
{
    const int wave = threadIdx.x >> 5;
    const int lane = threadIdx.x & 31;
    const int li   = lane & 15;
    const int hf   = lane >> 4;

    const int ntiles64 = N >> 6;
    const int task  = blockIdx.x * 8 + wave;       // wave-uniform
    const int mtile = task / ntiles64;
    const int ntile = task % ntiles64;
    if (mtile * 16 >= M) return;                   // wave-uniform exit (EXEC stays full)
    const int m0 = mtile * 16;
    const int n0 = ntile * 64;
    const int ntiles16 = N >> 4;

    v8f acc0 = {}, acc1 = {}, acc2 = {}, acc3 = {};

    const _Float16* ArowHi = Ahi + (long)(m0 + li) * Kp;
    const _Float16* ArowLo = Alo + (long)(m0 + li) * Kp;
    const int kt_end = Kp >> 5;

    for (int kt = 0; kt < kt_end; ++kt) {
        const int kb = kt << 5;
        // A fragment: elems 0..7 = K kb+8*hf+{0..7}; elems 8..15 = K kb+16+8*hf+{0..7}
        v8h ah0 = *(const v8h*)(ArowHi + kb + 8 * hf);
        v8h ah1 = *(const v8h*)(ArowHi + kb + 16 + 8 * hf);
        v8h al0 = *(const v8h*)(ArowLo + kb + 8 * hf);
        v8h al1 = *(const v8h*)(ArowLo + kb + 16 + 8 * hf);
        v16h ahi = __builtin_shufflevector(ah0, ah1, 0,1,2,3,4,5,6,7,8,9,10,11,12,13,14,15);
        v16h alo = __builtin_shufflevector(al0, al1, 0,1,2,3,4,5,6,7,8,9,10,11,12,13,14,15);

        const long tbase = ((long)kt * ntiles16 + (n0 >> 4)) * 512 + (long)lane * 16;
        {
            v16h bhi = *(const v16h*)(Bhi + tbase);
            v16h blo = *(const v16h*)(Blo + tbase);
            acc0 = __builtin_amdgcn_wmma_f32_16x16x32_f16(false, ahi, false, bhi, (short)0, acc0, false, false);
            acc0 = __builtin_amdgcn_wmma_f32_16x16x32_f16(false, ahi, false, blo, (short)0, acc0, false, false);
            acc0 = __builtin_amdgcn_wmma_f32_16x16x32_f16(false, alo, false, bhi, (short)0, acc0, false, false);
        }
        {
            v16h bhi = *(const v16h*)(Bhi + tbase + 512);
            v16h blo = *(const v16h*)(Blo + tbase + 512);
            acc1 = __builtin_amdgcn_wmma_f32_16x16x32_f16(false, ahi, false, bhi, (short)0, acc1, false, false);
            acc1 = __builtin_amdgcn_wmma_f32_16x16x32_f16(false, ahi, false, blo, (short)0, acc1, false, false);
            acc1 = __builtin_amdgcn_wmma_f32_16x16x32_f16(false, alo, false, bhi, (short)0, acc1, false, false);
        }
        {
            v16h bhi = *(const v16h*)(Bhi + tbase + 1024);
            v16h blo = *(const v16h*)(Blo + tbase + 1024);
            acc2 = __builtin_amdgcn_wmma_f32_16x16x32_f16(false, ahi, false, bhi, (short)0, acc2, false, false);
            acc2 = __builtin_amdgcn_wmma_f32_16x16x32_f16(false, ahi, false, blo, (short)0, acc2, false, false);
            acc2 = __builtin_amdgcn_wmma_f32_16x16x32_f16(false, alo, false, bhi, (short)0, acc2, false, false);
        }
        {
            v16h bhi = *(const v16h*)(Bhi + tbase + 1536);
            v16h blo = *(const v16h*)(Blo + tbase + 1536);
            acc3 = __builtin_amdgcn_wmma_f32_16x16x32_f16(false, ahi, false, bhi, (short)0, acc3, false, false);
            acc3 = __builtin_amdgcn_wmma_f32_16x16x32_f16(false, ahi, false, blo, (short)0, acc3, false, false);
            acc3 = __builtin_amdgcn_wmma_f32_16x16x32_f16(false, alo, false, bhi, (short)0, acc3, false, false);
        }
    }

    // D element (lane, v): row m0 + v + 8*hf, col n0 + 16*j + li
    #pragma unroll
    for (int v = 0; v < 8; ++v) {
        const int row = m0 + v + 8 * hf;
        float* crow = C + (long)row * N + n0 + li;
        float x0 = acc0[v] + bias[n0 + li];
        float x1 = acc1[v] + bias[n0 + 16 + li];
        float x2 = acc2[v] + bias[n0 + 32 + li];
        float x3 = acc3[v] + bias[n0 + 48 + li];
        if (mode == 0) {
            crow[0]  = fmaxf(x0, 0.0f);
            crow[16] = fmaxf(x1, 0.0f);
            crow[32] = fmaxf(x2, 0.0f);
            crow[48] = fmaxf(x3, 0.0f);
        } else if (mode == 1) {
            crow[0]  += fmaxf(x0, 0.0f);
            crow[16] += fmaxf(x1, 0.0f);
            crow[32] += fmaxf(x2, 0.0f);
            crow[48] += fmaxf(x3, 0.0f);
        } else {
            crow[0]  = x0;
            crow[16] = x1;
            crow[32] = x2;
            crow[48] = x3;
        }
    }
}

// ---------------- host orchestration ----------------

extern "C" void kernel_launch(void* const* d_in, const int* in_sizes, int n_in,
                              void* d_out, int out_size, void* d_ws, size_t ws_size,
                              hipStream_t stream) {
    (void)in_sizes; (void)n_in; (void)out_size; (void)ws_size;

    const float* feat_object = (const float*)d_in[1];
    const float* feat_room   = (const float*)d_in[2];
    const float* feat_attr   = (const float*)d_in[3];
    const float* W1_i = (const float*)d_in[4];
    const float* b1_i = (const float*)d_in[5];
    const float* W1_b = (const float*)d_in[6];
    const float* b1_b = (const float*)d_in[7];
    const float* W2   = (const float*)d_in[8];
    const float* b2   = (const float*)d_in[9];
    const int* src_oo = (const int*)d_in[10];
    const int* dst_oo = (const int*)d_in[11];
    const int* src_ro = (const int*)d_in[12];
    const int* dst_ro = (const int*)d_in[13];
    const int* src_ao = (const int*)d_in[14];
    const int* dst_ao = (const int*)d_in[15];

    float* out = (float*)d_out;
    float* ws  = (float*)d_ws;

    // small scratch in d_ws: degree/norm vectors (~330 KB)
    float* nsrc_obj  = ws;           // 20000
    float* nsrc_room = ws + 20000;   // 500
    float* nsrc_attr = ws + 20500;   // 2000
    float* ndst_oo   = ws + 22500;   // 20000
    float* ndst_ro   = ws + 42500;   // 20000
    float* ndst_ao   = ws + 62500;   // 20000  (end 82500)
    const long NORMS = 82500;

    // big scratch carved out of d_out's upper (to-be-broadcast) slices
    const long SL = (long)N_OBJ * HDIM;   // 10,240,000 floats per batch slice
    float* AGG  = out + 7 * SL;           // f32, up to 20000x512
    float* HOBJ = out + 6 * SL;           // f32, 20000x512
    float* AGG2 = out + 5 * SL;           // f32, 20000x512
    _Float16* Ahi = (_Float16*)(out + 4 * SL);          // up to 20000x512 f16
    _Float16* Alo = Ahi + (long)N_OBJ * HDIM;           // same size, fills slice 4
    _Float16* Bhi = (_Float16*)(out + 3 * SL);          // packed B fragments (<=512KB)
    _Float16* Blo = Bhi + (long)16 * 32 * 512;          // max (512/32)*(512/16)*512

    const int T = 256;
    auto blocks = [](long n, int t) { return (unsigned)((n + t - 1) / t); };

    // ---- degrees -> norms ----
    fill_kernel<<<blocks(NORMS, T), T, 0, stream>>>(ws, 0.0f, NORMS);
    degree_kernel<<<blocks(E_OO, T), T, 0, stream>>>(src_oo, dst_oo, nsrc_obj,  ndst_oo, E_OO);
    degree_kernel<<<blocks(E_RO, T), T, 0, stream>>>(src_ro, dst_ro, nsrc_room, ndst_ro, E_RO);
    degree_kernel<<<blocks(E_AO, T), T, 0, stream>>>(src_ao, dst_ao, nsrc_attr, ndst_ao, E_AO);
    norm_kernel<<<blocks(NORMS, T), T, 0, stream>>>(ws, NORMS);

    const int K1 = IN_F, Kp1 = 320;       // 300 padded to 32-multiple
    const int K2 = HDIM, Kp2 = HDIM;      // 512
    const long A1 = (long)N_OBJ * Kp1, A2 = (long)N_OBJ * Kp2;
    const long BP1 = (long)(Kp1 / 32) * (HDIM / 16) * 512;
    const long BP2 = (long)(Kp2 / 32) * (HDIM / 16) * 512;
    const long AGG1_N = (long)N_OBJ * IN_F;
    const long AGG2_N = (long)N_OBJ * HDIM;
    const unsigned gemm_grid = (N_OBJ / 16) * (HDIM / 64) / 8; // 1250 blocks

    // ---- conv1: object->object ----
    fill_kernel<<<blocks(AGG1_N, T), T, 0, stream>>>(AGG, 0.0f, AGG1_N);
    scatter_kernel<<<E_OO, T, 0, stream>>>(feat_object, src_oo, dst_oo, nsrc_obj, AGG, IN_F);
    convA_kernel<<<blocks(A1, T), T, 0, stream>>>(AGG, ndst_oo, 1.0f, Ahi, Alo, K1, Kp1, A1);
    convB_kernel<<<blocks(BP1, T), T, 0, stream>>>(W1_i, Bhi, Blo, K1, HDIM, BP1);
    gemm_wmma_f16x2<<<gemm_grid, T, 0, stream>>>(Ahi, Alo, Kp1, Bhi, Blo, b1_i, HOBJ, N_OBJ, HDIM, 0);

    // ---- conv1: room->object ----
    fill_kernel<<<blocks(AGG1_N, T), T, 0, stream>>>(AGG, 0.0f, AGG1_N);
    scatter_kernel<<<E_RO, T, 0, stream>>>(feat_room, src_ro, dst_ro, nsrc_room, AGG, IN_F);
    convA_kernel<<<blocks(A1, T), T, 0, stream>>>(AGG, ndst_ro, 1.0f, Ahi, Alo, K1, Kp1, A1);
    convB_kernel<<<blocks(BP1, T), T, 0, stream>>>(W1_i, Bhi, Blo, K1, HDIM, BP1);
    gemm_wmma_f16x2<<<gemm_grid, T, 0, stream>>>(Ahi, Alo, Kp1, Bhi, Blo, b1_i, HOBJ, N_OBJ, HDIM, 1);

    // ---- conv1: attr->object ----
    fill_kernel<<<blocks(AGG1_N, T), T, 0, stream>>>(AGG, 0.0f, AGG1_N);
    scatter_kernel<<<E_AO, T, 0, stream>>>(feat_attr, src_ao, dst_ao, nsrc_attr, AGG, IN_F);
    convA_kernel<<<blocks(A1, T), T, 0, stream>>>(AGG, ndst_ao, 1.0f, Ahi, Alo, K1, Kp1, A1);
    convB_kernel<<<blocks(BP1, T), T, 0, stream>>>(W1_b, Bhi, Blo, K1, HDIM, BP1);
    gemm_wmma_f16x2<<<gemm_grid, T, 0, stream>>>(Ahi, Alo, Kp1, Bhi, Blo, b1_b, HOBJ, N_OBJ, HDIM, 1);
    // HOBJ = h_oo + h_ro + h_ao ; the /3 is folded into conv2's convA scale.

    // ---- conv2: object->object on h_obj ----
    fill_kernel<<<blocks(AGG2_N, T), T, 0, stream>>>(AGG2, 0.0f, AGG2_N);
    scatter_kernel<<<E_OO, T, 0, stream>>>(HOBJ, src_oo, dst_oo, nsrc_obj, AGG2, HDIM);
    convA_kernel<<<blocks(A2, T), T, 0, stream>>>(AGG2, ndst_oo, 1.0f / 3.0f, Ahi, Alo, K2, Kp2, A2);
    convB_kernel<<<blocks(BP2, T), T, 0, stream>>>(W2, Bhi, Blo, K2, HDIM, BP2);
    gemm_wmma_f16x2<<<gemm_grid, T, 0, stream>>>(Ahi, Alo, Kp2, Bhi, Blo, b2, out, N_OBJ, HDIM, 2);

    // ---- broadcast over frame batch (overwrites all scratch slices last) ----
    bcast_kernel<<<blocks(SL, T), T, 0, stream>>>(out, SL, NBATCH);
}
